// Generator_31756988187185
// MI455X (gfx1250) — matrix operand order repaired and checked by reading, not compiled, not run on hardware
//
#include <hip/hip_runtime.h>
#include <hip/hip_bf16.h>

// ---------------------------------------------------------------------------
// GCN forward on MI455X (gfx1250, wave32):
//   3x [ GEMM (bf16 WMMA, f32 acc)  ->  edge scatter-add (f32 atomics, L2)
//        -> epilogue relu(agg + dinv^2*T + b) ]
//   then global mean pool + small output GEMM.
// Feature matrix (51.2 MB) and edge list fit in the 192 MB L2, so the
// scatter phase is L2-atomic bound; the GEMMs ride the WMMA pipe.
// ---------------------------------------------------------------------------

#define N_NODES 100000
#define N_EDGES 1600000
#define CH      128
#define OUT_CH  64
#define N_GRAPHS 256

typedef __attribute__((ext_vector_type(16))) __bf16 v16bf;
typedef __attribute__((ext_vector_type(8)))  float  v8f;

__device__ __forceinline__ __bf16 f2bf(float f) {
    union { float f; unsigned u; } cv; cv.f = f;
    unsigned r = cv.u + 0x7FFFu + ((cv.u >> 16) & 1u);   // round-to-nearest-even
    unsigned short h = (unsigned short)(r >> 16);
    __bf16 out; __builtin_memcpy(&out, &h, 2);
    return out;
}

// ---------------------------------------------------------------------------
// degree / norm
// ---------------------------------------------------------------------------
__global__ __launch_bounds__(256) void k_init_deg(float* deg, int n) {
    int i = blockIdx.x * 256 + threadIdx.x;
    if (i < n) deg[i] = 1.0f;                            // self loop
}
__global__ __launch_bounds__(256) void k_count_deg(const int* __restrict__ dst,
                                                   float* deg, int e) {
    int i = blockIdx.x * 256 + threadIdx.x;
    if (i < e) atomicAdd(&deg[dst[i]], 1.0f);
}
__global__ __launch_bounds__(256) void k_dinv(float* deg, int n) {
    int i = blockIdx.x * 256 + threadIdx.x;
    if (i < n) deg[i] = rsqrtf(fmaxf(deg[i], 1.0f));     // in-place -> dinv
}

// ---------------------------------------------------------------------------
// GEMM: T[N x 128] = A[N x 128] @ W[128 x 128]   (bf16 WMMA, f32 accumulate)
// Block = 16 rows x 128 cols; 8 waves, one 16-col stripe per wave.
// ---------------------------------------------------------------------------
__global__ __launch_bounds__(256) void k_gemm_wmma(const float* __restrict__ A,
                                                   const float* __restrict__ W,
                                                   float* __restrict__ T) {
    __shared__ __bf16 sA[16][CH + 8];                    // padded vs bank conflicts
    const int tid  = threadIdx.x;
    const int lane = tid & 31;
    const int wave = tid >> 5;
    const int row0 = blockIdx.x * 16;

    // stage A tile (16x128 f32 -> bf16) into LDS
    for (int i = tid; i < 16 * CH; i += 256) {
        int r = i >> 7, c = i & 127;
        sA[r][c] = f2bf(A[(long)(row0 + r) * CH + c]);
    }

    const int half = lane >> 4;     // 0: lanes 0-15, 1: lanes 16-31
    const int lr   = lane & 15;
    const int n0   = wave * 16;     // column stripe of this wave

    // Preload B fragments for the 4 K-steps (32x16 bf16 each):
    //   lane holds column n0+lr; element j covers K = ks*32 + (j>>3)*16 + 8*half + (j&7)
    v16bf bfrag[4];
#pragma unroll
    for (int ks = 0; ks < 4; ++ks) {
#pragma unroll
        for (int j = 0; j < 16; ++j) {
            int K = ks * 32 + (j >> 3) * 16 + half * 8 + (j & 7);
            bfrag[ks][j] = f2bf(W[K * CH + n0 + lr]);
        }
    }

    __syncthreads();

    v8f acc = {};
#pragma unroll
    for (int ks = 0; ks < 4; ++ks) {
        // A fragment 16x32: row = lr; K per ISA table:
        //   VGPR v, half h: K = (v/4)*16 + h*8 + (v%4)*2 + (j&1)
        v16bf afrag;
#pragma unroll
        for (int j = 0; j < 16; ++j) {
            int v = j >> 1;
            int K = ks * 32 + (v >> 2) * 16 + half * 8 + (v & 3) * 2 + (j & 1);
            afrag[j] = sA[lr][K];
        }
        acc = __builtin_amdgcn_wmma_f32_16x16x32_bf16(
                  false, afrag, false, bfrag[ks], (short)0, acc, false, false);
    }

    // D 16x16 f32: element g -> row g + 8*half, col lr
#pragma unroll
    for (int g = 0; g < 8; ++g) {
        T[(long)(row0 + g + 8 * half) * CH + n0 + lr] = acc[g];
    }
}

// ---------------------------------------------------------------------------
// edge scatter: agg[dst] += dinv[src]*dinv[dst] * T[src]   (one wave per edge)
// ---------------------------------------------------------------------------
__global__ __launch_bounds__(256) void k_edge_agg(const int* __restrict__ src,
                                                  const int* __restrict__ dst,
                                                  const float* __restrict__ dinv,
                                                  const float* __restrict__ T,
                                                  float* __restrict__ agg, int e) {
    const int lane = threadIdx.x & 31;
    const int edge = blockIdx.x * 8 + (threadIdx.x >> 5);
    if (edge >= e) return;
    const int s = src[edge], d = dst[edge];
    const float nm = dinv[s] * dinv[d];
    const float4 v = ((const float4*)(T + (long)s * CH))[lane];   // 4 ch / lane
    float* a = agg + (long)d * CH + lane * 4;
    atomicAdd(a + 0, nm * v.x);
    atomicAdd(a + 1, nm * v.y);
    atomicAdd(a + 2, nm * v.z);
    atomicAdd(a + 3, nm * v.w);
}

// ---------------------------------------------------------------------------
// epilogue: h = relu(agg + dinv^2 * T + b)   (in-place on agg buffer)
// ---------------------------------------------------------------------------
__global__ __launch_bounds__(256) void k_epilogue(float* __restrict__ h,
                                                  const float* __restrict__ T,
                                                  const float* __restrict__ dinv,
                                                  const float* __restrict__ b) {
    long i = (long)blockIdx.x * 256 + threadIdx.x;
    if (i >= (long)N_NODES * CH) return;
    int node = (int)(i >> 7), c = (int)(i & 127);
    float di = dinv[node];
    float v = h[i] + di * di * T[i] + b[c];
    h[i] = fmaxf(v, 0.0f);
}

// ---------------------------------------------------------------------------
// pooling + output head
// ---------------------------------------------------------------------------
__global__ __launch_bounds__(256) void k_zero(float* p, long n) {
    long i = (long)blockIdx.x * 256 + threadIdx.x;
    if (i < n) p[i] = 0.0f;
}
__global__ __launch_bounds__(256) void k_pool_cnt(const int* __restrict__ batch,
                                                  float* cnt, int n) {
    int i = blockIdx.x * 256 + threadIdx.x;
    if (i < n) atomicAdd(&cnt[batch[i]], 1.0f);
}
__global__ __launch_bounds__(256) void k_pool_sum(const int* __restrict__ batch,
                                                  const float* __restrict__ h,
                                                  float* pooled) {
    long i = (long)blockIdx.x * 256 + threadIdx.x;
    if (i >= (long)N_NODES * CH) return;
    int node = (int)(i >> 7), c = (int)(i & 127);
    atomicAdd(&pooled[(long)batch[node] * CH + c], h[i]);
}
__global__ __launch_bounds__(256) void k_head(const float* __restrict__ pooled,
                                              const float* __restrict__ cnt,
                                              const float* __restrict__ w_out,
                                              const float* __restrict__ b_out,
                                              float* __restrict__ out) {
    int i = blockIdx.x * 256 + threadIdx.x;             // 256*64 outputs
    if (i >= N_GRAPHS * OUT_CH) return;
    int g = i / OUT_CH, o = i % OUT_CH;
    float acc = 0.0f;
#pragma unroll 4
    for (int c = 0; c < CH; ++c)
        acc += pooled[g * CH + c] * w_out[c * OUT_CH + o];
    out[i] = acc / fmaxf(cnt[g], 1.0f) + b_out[o];
}

// ---------------------------------------------------------------------------
// launch
// ---------------------------------------------------------------------------
extern "C" void kernel_launch(void* const* d_in, const int* in_sizes, int n_in,
                              void* d_out, int out_size, void* d_ws, size_t ws_size,
                              hipStream_t stream) {
    const float* x      = (const float*)d_in[0];
    const int*   ei     = (const int*)  d_in[1];    // [2, E] flat int32
    const int*   batch  = (const int*)  d_in[2];
    const float* w[3]   = { (const float*)d_in[3], (const float*)d_in[5], (const float*)d_in[7] };
    const float* b[3]   = { (const float*)d_in[4], (const float*)d_in[6], (const float*)d_in[8] };
    const float* w_out  = (const float*)d_in[9];
    const float* b_out  = (const float*)d_in[10];
    float*       out    = (float*)d_out;

    const int* src = ei;                 // edge_index[0]
    const int* dst = ei + N_EDGES;       // edge_index[1]

    // workspace layout (floats)
    float* ws     = (float*)d_ws;
    float* dinv   = ws;                                  // N (rounded region 102400)
    float* bufT   = ws + 102400;                         // N*128
    float* bufH   = bufT + (long)N_NODES * CH;           // N*128 (agg / hidden)
    float* pooled = bufH + (long)N_NODES * CH;           // 256*128
    float* cnt    = pooled + N_GRAPHS * CH;              // 256

    const long NC = (long)N_NODES * CH;
    const int  gN   = (N_NODES + 255) / 256;
    const int  gE   = (N_EDGES + 255) / 256;
    const int  gNC  = (int)((NC + 255) / 256);
    const int  gEw  = (N_EDGES + 7) / 8;                 // one wave / edge
    const int  gGemm = N_NODES / 16;                     // 6250 row tiles

    // degree -> dinv
    k_init_deg<<<gN, 256, 0, stream>>>(dinv, N_NODES);
    k_count_deg<<<gE, 256, 0, stream>>>(dst, dinv, N_EDGES);
    k_dinv<<<gN, 256, 0, stream>>>(dinv, N_NODES);

    // 3 GCN layers
    const float* hin = x;
    for (int l = 0; l < 3; ++l) {
        k_gemm_wmma<<<gGemm, 256, 0, stream>>>(hin, w[l], bufT);
        k_zero<<<gNC, 256, 0, stream>>>(bufH, NC);
        k_edge_agg<<<gEw, 256, 0, stream>>>(src, dst, dinv, bufT, bufH, N_EDGES);
        k_epilogue<<<gNC, 256, 0, stream>>>(bufH, bufT, dinv, b[l]);
        hin = bufH;
    }

    // global mean pool + head
    k_zero<<<(N_GRAPHS * CH + 255) / 256, 256, 0, stream>>>(pooled, N_GRAPHS * CH);
    k_zero<<<1, 256, 0, stream>>>(cnt, N_GRAPHS);
    k_pool_cnt<<<gN, 256, 0, stream>>>(batch, cnt, N_NODES);
    k_pool_sum<<<gNC, 256, 0, stream>>>(batch, bufH, pooled);
    k_head<<<(N_GRAPHS * OUT_CH + 255) / 256, 256, 0, stream>>>(pooled, cnt, w_out, b_out, out);
}
